// Attention_13709535609283
// MI455X (gfx1250) — matrix operand order repaired and checked by reading, not compiled, not run on hardware
//
#include <hip/hip_runtime.h>
#include <hip/hip_bf16.h>

// ---------------------------------------------------------------------------
// MI455X (gfx1250) implementation.
//   B=8, S=256, C=64, H=256.  Phases:
//   K1 conv      : p,u,v = conv1d(h; w0/w1a/w1b)          (VALU, streams h)
//   K2 gemm      : Ai,Bj  = p/u/v @ fc*^T + biases        (bf16 WMMA, f32 acc)
//   K3 scan      : 256-step tanh recurrence, rows indep.  (bf16 WMMA, f32 acc)
//   K4 norm      : relu/sigmoid/softmax -> probs in-place (VALU + prefetch)
//   K5 ctx       : context = probs x h  (per (b,c) 256^3) (TDM->LDS + bf16 WMMA)
//   K6 copy      : out[:, S:2S] = h                       (float4 copy)
// Workspace: ~142 MB of f32 scratch in d_ws.
// ---------------------------------------------------------------------------

typedef __attribute__((ext_vector_type(16))) __bf16 v16bf;
typedef __attribute__((ext_vector_type(8)))  float  v8f;
typedef __attribute__((ext_vector_type(4)))  unsigned int v4u;
typedef __attribute__((ext_vector_type(8)))  int v8i;
typedef __attribute__((ext_vector_type(4)))  int v4i;

#define B_ 8
#define S_ 256
#define C_ 64
#define H_ 256
#define N_ (B_*S_)   // 2048 rows

#define HAVE_TDM (__has_builtin(__builtin_amdgcn_tensor_load_to_lds) && \
                  __has_builtin(__builtin_amdgcn_s_wait_tensorcnt))

#if HAVE_TDM
// Build a minimal 2D-tile D# (groups 0/1 per CDNA5 ISA)
// and issue TENSOR_LOAD_TO_LDS. tile_d0 elements (f32) per row, tile_d1 rows,
// row stride stride0 elements. tensor dims == tile dims (no OOB region needed).
__device__ __forceinline__ void tdm_load_tile_2d(unsigned int lds_off, const void* gptr,
                                                 unsigned int tile_d0, unsigned int tile_d1,
                                                 unsigned int stride0_elts)
{
  unsigned long long ga = (unsigned long long)(uintptr_t)gptr;
  v4u g0;
  g0.x = 1u;                                       // count=1, is_restore=0, no gather
  g0.y = lds_off;                                  // lds_addr (offset in LDS)
  g0.z = (unsigned int)ga;                         // global_addr[31:0]
  g0.w = (unsigned int)(ga >> 32) | (2u << 30);    // global_addr[56:32] | type=2
  v8i g1;
  g1[0] = (int)(2u << 16);                         // data_size=2 -> 4 bytes
  g1[1] = (int)((tile_d0 & 0xFFFFu) << 16);        // tensor_dim0[15:0]
  g1[2] = (int)((tile_d1 & 0xFFFFu) << 16);        // tensor_dim0 hi=0 | tensor_dim1 lo
  g1[3] = (int)((tile_d0 & 0xFFFFu) << 16);        // tensor_dim1 hi=0 | tile_dim0
  g1[4] = (int)(tile_d1 & 0xFFFFu);                // tile_dim1 | tile_dim2=0
  g1[5] = (int)stride0_elts;                       // tensor_dim0_stride[31:0]
  g1[6] = 0;
  g1[7] = 0;
  v4i z4 = {0, 0, 0, 0};
#if __clang_major__ >= 23
  v8i z8 = {0, 0, 0, 0, 0, 0, 0, 0};
  __builtin_amdgcn_tensor_load_to_lds(g0, g1, z4, z4, z8, 0);
#else
  __builtin_amdgcn_tensor_load_to_lds(g0, g1, z4, z4, 0);
#endif
}
#endif

// ---------------------------------------------------------------- K1: conv1d
__global__ __launch_bounds__(256) void conv_kernel(
    const float* __restrict__ h,
    const float* __restrict__ w0, const float* __restrict__ b0,
    const float* __restrict__ w1,
    float* __restrict__ p, float* __restrict__ u, float* __restrict__ v)
{
  int n  = blockIdx.x;      // row in [0, 2048)
  int hp = threadIdx.x;     // output position in [0, 256)
  const float* x = h + (size_t)n * C_ * H_;
  float a0 = 0.f, a1 = 0.f, a2 = 0.f;
  for (int c = 0; c < C_; ++c) {
    const float* xr  = x  + c * H_;
    const float* wr0 = w0 + c * 9;
    const float* wr1 = w1 + c * 9;          // w1[0]
    const float* wr2 = w1 + 576 + c * 9;    // w1[1]
    #pragma unroll
    for (int k = 0; k < 9; ++k) {
      int idx = hp - 4 + k;
      float xv = (idx >= 0 && idx < H_) ? xr[idx] : 0.f;
      a0 = fmaf(xv, wr0[k], a0);
      a1 = fmaf(xv, wr1[k], a1);
      a2 = fmaf(xv, wr2[k], a2);
    }
  }
  size_t o = (size_t)n * H_ + hp;
  p[o] = a0 + b0[0];
  u[o] = a1;
  v[o] = a2;
}

// ------------------------------------------------------- K2: Ai/Bj via WMMA
// Ai = p@fc00^T + u@fc1^T + (fc00_b + fc1_b + b1*rowsum(fc1_w))
// Bj = p@fc01^T + v@fc1^T + fc01_b
__global__ __launch_bounds__(32) void gemm_aibj_kernel(
    const float* __restrict__ p, const float* __restrict__ u, const float* __restrict__ v,
    const float* __restrict__ fc00_w, const float* __restrict__ fc00_b,
    const float* __restrict__ fc01_w, const float* __restrict__ fc01_b,
    const float* __restrict__ fc1_w,  const float* __restrict__ fc1_b,
    const float* __restrict__ b1,
    float* __restrict__ Ai, float* __restrict__ Bj)
{
  int tile = blockIdx.x;                 // 512 = (2048/16) row-tiles * (64/16) col-tiles
  int ct = tile & 3, rt = tile >> 2;
  int lane = threadIdx.x;
  int ln = lane & 15, g = lane >> 4;
  int row0 = rt * 16, col0 = ct * 16;
  int m = ln;                            // A-fragment row for this lane
  int n = col0 + ln;                     // output column for this lane
  v8f accA = {};
  v8f accB = {};
  for (int kb = 0; kb < H_; kb += 32) {
    v16bf aP, aU, aV, b00, b01, b1f;
    #pragma unroll
    for (int i = 0; i < 16; ++i) {
      int kA = kb + 16 * (i >> 3) + 8 * g + (i & 7);   // 16-bit A layout
      size_t ao = (size_t)(row0 + m) * H_ + kA;
      aP[i] = (__bf16)p[ao];
      aU[i] = (__bf16)u[ao];
      aV[i] = (__bf16)v[ao];
      int kB = kb + 16 * (i >> 3) + 8 * g + (i & 7);   // 16-bit B layout
      size_t bo = (size_t)n * H_ + kB;                 // B[k][n] = W[n][k]
      b00[i] = (__bf16)fc00_w[bo];
      b01[i] = (__bf16)fc01_w[bo];
      b1f[i] = (__bf16)fc1_w[bo];
    }
    accA = __builtin_amdgcn_wmma_f32_16x16x32_bf16(false, aP, false, b00, (short)0, accA, false, false);
    accA = __builtin_amdgcn_wmma_f32_16x16x32_bf16(false, aU, false, b1f, (short)0, accA, false, false);
    accB = __builtin_amdgcn_wmma_f32_16x16x32_bf16(false, aP, false, b01, (short)0, accB, false, false);
    accB = __builtin_amdgcn_wmma_f32_16x16x32_bf16(false, aV, false, b1f, (short)0, accB, false, false);
  }
  float rs = 0.f;
  for (int k = 0; k < H_; ++k) rs += fc1_w[(size_t)n * H_ + k];
  float biasA = fc00_b[n] + fc1_b[n] + b1[0] * rs;
  float biasB = fc01_b[n];
  #pragma unroll
  for (int r = 0; r < 8; ++r) {
    int mm = row0 + r + 8 * g;
    Ai[(size_t)mm * C_ + n] = accA[r] + biasA;
    Bj[(size_t)mm * C_ + n] = accB[r] + biasB;
  }
}

// -------------------------------------------------- K3: sequential tanh scan
// One wave owns 16 rows; state s[16][64] lives in LDS; fc2^T resident in VGPRs.
__global__ __launch_bounds__(32) void scan_kernel(
    const float* __restrict__ Ai, const float* __restrict__ Bj,
    const float* __restrict__ fc2_w, const float* __restrict__ fc2_b,
    float* __restrict__ scores)
{
  __shared__ float s[16 * C_];
  int rt = blockIdx.x;                   // 128 row tiles of 16
  int lane = threadIdx.x, ln = lane & 15, g = lane >> 4;
  int row0 = rt * 16;
  int bb = row0 / S_;                    // batch (tile never straddles batches)

  // B fragments of fc2^T: B[k][n] = fc2_w[n*C + k]; K=64 -> 2 ktiles, N=64 -> 4 ntiles
  v16bf bf2[2][4];
  #pragma unroll
  for (int kt = 0; kt < 2; ++kt)
    #pragma unroll
    for (int nt = 0; nt < 4; ++nt) {
      int n = nt * 16 + ln;
      #pragma unroll
      for (int i = 0; i < 16; ++i) {
        int k = 32 * kt + 16 * (i >> 3) + 8 * g + (i & 7);
        bf2[kt][nt][i] = (__bf16)fc2_w[(size_t)n * C_ + k];
      }
    }

  // Hoist Ai + fc2_b (loop-invariant per row-tile)
  float aib[4][8];
  #pragma unroll
  for (int nt = 0; nt < 4; ++nt)
    #pragma unroll
    for (int r = 0; r < 8; ++r) {
      int n = nt * 16 + ln;
      int mm = row0 + r + 8 * g;
      aib[nt][r] = Ai[(size_t)mm * C_ + n] + fc2_b[n];
    }

  for (int t = lane; t < 16 * C_; t += 32) s[t] = 0.f;
  __syncthreads();

  for (int j = 0; j < S_; ++j) {
    v16bf a0, a1;
    #pragma unroll
    for (int i = 0; i < 16; ++i) {
      int k32 = 16 * (i >> 3) + 8 * g + (i & 7);
      a0[i] = (__bf16)s[ln * C_ + k32];
      a1[i] = (__bf16)s[ln * C_ + 32 + k32];
    }
    float bjv[4];
    #pragma unroll
    for (int nt = 0; nt < 4; ++nt)
      bjv[nt] = Bj[((size_t)bb * S_ + j) * C_ + nt * 16 + ln];
    __syncthreads();
    #pragma unroll
    for (int nt = 0; nt < 4; ++nt) {
      v8f acc = {};
      acc = __builtin_amdgcn_wmma_f32_16x16x32_bf16(false, a0, false, bf2[0][nt], (short)0, acc, false, false);
      acc = __builtin_amdgcn_wmma_f32_16x16x32_bf16(false, a1, false, bf2[1][nt], (short)0, acc, false, false);
      #pragma unroll
      for (int r = 0; r < 8; ++r) {
        int n = nt * 16 + ln;
        int mloc = r + 8 * g;
        float val = tanhf(aib[nt][r] + bjv[nt] + acc[r]);
        scores[(((size_t)(row0 + mloc)) * S_ + j) * C_ + n] = val;  // [b,i,j,c]
        s[mloc * C_ + n] = val;
      }
    }
    __syncthreads();
  }
}

// -------------------------------------- K4: gd / sigmoid-mix / masked softmax
// One block per (b,i); thread = j.  probs overwrite scores in place.
__global__ __launch_bounds__(256) void norm_kernel(float* __restrict__ scores)
{
  __shared__ float red[256];
  int bi = blockIdx.x;                   // b*S + i
  int i  = bi & (S_ - 1);
  int j  = threadIdx.x;
  bool valid = (j < i);
  size_t base = ((size_t)bi * S_ + j) * C_;

  // Pull this thread's 64-float run into near caches; we re-read it 4x below.
  #pragma unroll
  for (int cc = 0; cc < C_; cc += 32) __builtin_prefetch(&scores[base + cc], 0, 3);

  float grel = 0.f, gsig = 0.f;
  for (int c = 0; c < C_; ++c) {
    float x = scores[base + c];
    grel += fmaxf(x, 0.f);
    gsig += 1.f / (1.f + __expf(-x));
  }
  float gdv = valid ? grel : 0.f;
  float inv_ss = 1.f / (gsig + 1e-8f);

  red[j] = gdv; __syncthreads();
  for (int st = 128; st > 0; st >>= 1) { if (j < st) red[j] += red[j + st]; __syncthreads(); }
  float gd = gdv / (red[0] + 1e-8f); __syncthreads();

  float lm = -1e9f;
  if (valid) {
    for (int c = 0; c < C_; ++c) {
      float x = scores[base + c];
      float sig = 1.f / (1.f + __expf(-x));
      lm = fmaxf(lm, 2.0f * gd * sig * inv_ss);
    }
  }
  red[j] = lm; __syncthreads();
  for (int st = 128; st > 0; st >>= 1) { if (j < st) red[j] = fmaxf(red[j], red[j + st]); __syncthreads(); }
  float mx = red[0]; __syncthreads();

  float es = 0.f;
  if (valid) {
    for (int c = 0; c < C_; ++c) {
      float x = scores[base + c];
      float sig = 1.f / (1.f + __expf(-x));
      es += __expf(2.0f * gd * sig * inv_ss - mx);
    }
  }
  red[j] = es; __syncthreads();
  for (int st = 128; st > 0; st >>= 1) { if (j < st) red[j] += red[j + st]; __syncthreads(); }
  float denom = red[0];
  float invd = (denom > 0.f) ? 1.f / denom : 0.f;   // i==0 row -> all zeros

  for (int c = 0; c < C_; ++c) {
    float pr = 0.f;
    if (valid) {
      float x = scores[base + c];
      float sig = 1.f / (1.f + __expf(-x));
      pr = __expf(2.0f * gd * sig * inv_ss - mx) * invd;
    }
    scores[base + c] = pr;
  }
}

// ----------------------------------------------- K5: context einsum via WMMA
// Per (b,c): D[i][hh] = sum_j probs[b,i,j,c] * h[b,j,c,hh]  (256x256x256 GEMM).
// 128-thread block per (b, c, 16-row i-tile). The h chunk [32 j][256 hh]
// (row stride C*H*4 = 64 KB) is DMA'd into LDS by the Tensor Data Mover
// (tracked with TENSORcnt); the probs tile [16 i][32 j] (stride-C gather) is
// staged cooperatively. 4 waves each own 4 column tiles of the output row.
__global__ __launch_bounds__(128) void ctx_kernel(
    const float* __restrict__ probs,    // scores buffer now holds probs
    const float* __restrict__ h,
    float* __restrict__ out)
{
  __shared__ __align__(16) float Bs[32][256];   // h chunk  [j][hh], 32 KB
  __shared__ float As[16][32];                  // probs tile [i][j], 2 KB
  int bid = blockIdx.x;                 // 8*64*16 = 8192 blocks
  int mt = bid & 15;
  int c  = (bid >> 4) & 63;
  int b  = bid >> 10;
  int tid = threadIdx.x;
  int wv = tid >> 5;                    // wave 0..3 -> column tiles 4wv..4wv+3
  int lane = tid & 31, ln = lane & 15, g = lane >> 4;
  int i0 = mt * 16;
  v8f acc[4] = {{}, {}, {}, {}};

  for (int kb = 0; kb < 8; ++kb) {
    int j0 = kb * 32;
    // ---- stage h chunk -> Bs (TDM 2D tile load if available)
#if HAVE_TDM
    if (tid < 32) {                     // wave 0 issues the DMA (EXEC-independent)
      const void* gsrc = (const void*)(h + (((size_t)b * S_ + j0) * C_ + c) * H_);
      tdm_load_tile_2d((unsigned int)(uintptr_t)&Bs[0][0], gsrc,
                       /*tile_d0=*/H_, /*tile_d1=*/32, /*stride0=*/C_ * H_);
      __builtin_amdgcn_s_wait_tensorcnt(0);
    }
#else
    for (int t = tid; t < 32 * 64; t += 128) {   // float4 granularity
      int row = t >> 6, c4 = t & 63;
      const float4* src = (const float4*)(h + (((size_t)b * S_ + j0 + row) * C_ + c) * H_) + c4;
      *(float4*)&Bs[row][c4 * 4] = *src;
    }
#endif
    // ---- stage probs tile -> As (stride-C gather, every 32-j step)
    for (int t = tid; t < 16 * 32; t += 128) {
      int ii = t >> 5, jj = t & 31;
      As[ii][jj] = probs[(((size_t)b * S_ + i0 + ii) * S_ + j0 + jj) * C_ + c];
    }
    __syncthreads();
    // ---- compute: one A fragment shared across this wave's 4 column tiles
    v16bf av;
    #pragma unroll
    for (int i = 0; i < 16; ++i)
      av[i] = (__bf16)As[ln][16 * (i >> 3) + 8 * g + (i & 7)];
    #pragma unroll
    for (int q = 0; q < 4; ++q) {
      int col0 = (wv * 4 + q) * 16;
      v16bf bv;
      #pragma unroll
      for (int i = 0; i < 16; ++i)
        bv[i] = (__bf16)Bs[16 * (i >> 3) + 8 * g + (i & 7)][col0 + ln];
      acc[q] = __builtin_amdgcn_wmma_f32_16x16x32_bf16(false, av, false, bv, (short)0, acc[q], false, false);
    }
    __syncthreads();
  }
  #pragma unroll
  for (int q = 0; q < 4; ++q) {
    int col0 = (wv * 4 + q) * 16;
    #pragma unroll
    for (int r = 0; r < 8; ++r) {
      int i = i0 + r + 8 * g;
      out[(((size_t)b * 2 * S_ + i) * C_ + c) * H_ + col0 + ln] = acc[q][r];
    }
  }
}

// ------------------------------------------------------ K6: out[:,S:2S] = h
__global__ __launch_bounds__(256) void copy_h_kernel(
    const float* __restrict__ h, float* __restrict__ out)
{
  size_t q = (size_t)blockIdx.x * blockDim.x + threadIdx.x;  // float4 index
  const size_t SCH4 = (size_t)S_ * C_ * H_ / 4;              // 1048576
  size_t b = q / SCH4;
  const float4* src = (const float4*)h;
  float4* dst = (float4*)out;
  dst[q + (b + 1) * SCH4] = src[q];
}

// ---------------------------------------------------------------------------
extern "C" void kernel_launch(void* const* d_in, const int* in_sizes, int n_in,
                              void* d_out, int out_size, void* d_ws, size_t ws_size,
                              hipStream_t stream)
{
  const float* h      = (const float*)d_in[0];
  const float* w0     = (const float*)d_in[1];
  const float* b0     = (const float*)d_in[2];
  const float* w1     = (const float*)d_in[3];
  const float* b1     = (const float*)d_in[4];
  const float* fc00_w = (const float*)d_in[5];
  const float* fc00_b = (const float*)d_in[6];
  const float* fc01_w = (const float*)d_in[7];
  const float* fc01_b = (const float*)d_in[8];
  const float* fc1_w  = (const float*)d_in[9];
  const float* fc1_b  = (const float*)d_in[10];
  const float* fc2_w  = (const float*)d_in[11];
  const float* fc2_b  = (const float*)d_in[12];
  float* out = (float*)d_out;

  // Workspace layout (floats): p | u | v | Ai | Bj | scores  (~142 MB total)
  float* w  = (float*)d_ws;
  float* p  = w;                       // 2048*256
  float* u  = w + 524288;
  float* v  = w + 1048576;
  float* Ai = w + 1572864;             // 2048*64
  float* Bj = w + 1703936;
  float* sc = w + 1835008;             // 8*256*256*64 = 33554432

  conv_kernel<<<N_, 256, 0, stream>>>(h, w0, b0, w1, p, u, v);
  gemm_aibj_kernel<<<(N_ / 16) * (C_ / 16), 32, 0, stream>>>(
      p, u, v, fc00_w, fc00_b, fc01_w, fc01_b, fc1_w, fc1_b, b1, Ai, Bj);
  scan_kernel<<<N_ / 16, 32, 0, stream>>>(Ai, Bj, fc2_w, fc2_b, sc);
  norm_kernel<<<N_, 256, 0, stream>>>(sc);
  ctx_kernel<<<B_ * C_ * (S_ / 16), 128, 0, stream>>>(sc, h, out);
  copy_h_kernel<<<(B_ * S_ * C_ * H_ / 4) / 256, 256, 0, stream>>>(h, out);
}
